// IndependentNodeNetwork2_53549652247126
// MI455X (gfx1250) — matrix operand order repaired and checked by reading, not compiled, not run on hardware
//
#include <hip/hip_runtime.h>
#include <hip/hip_bf16.h>

// Problem constants (from reference): B=8192 rows, N=64 nodes, C=256
#define BDIM   8192
#define NNODES 64
#define CDIM   256

typedef __attribute__((ext_vector_type(16))) _Float16 v16h;
typedef __attribute__((ext_vector_type(8)))  float    v8f;

union AFrag { v16h v; _Float16 h[16]; };
union BFrag { v16h v; unsigned int u[8]; };

// LDS panel row stride in bytes: 32 f16 = 64B data + 8B pad -> 72B.
// 72/4 = 18 banks/row; 18*d mod 64 distinct over 16-32 consecutive d
// -> (near) conflict-free ds_load b32/b64 B-fragment gathers.
#define PANEL_STRIDE 72
#define PANEL_BYTES  (CDIM * PANEL_STRIDE)

// ---------------------------------------------------------------------------
// Kernel 1: convert+transpose W2[n][k][d] (f32) -> W2T[n][d][k] (f16).
// One-time ~17MB pass; negligible vs the 68.7 GFLOP GEMM.
// ---------------------------------------------------------------------------
__global__ void prep_w2_kernel(const float* __restrict__ W2,
                               _Float16* __restrict__ W2T) {
    int idx = blockIdx.x * 256 + threadIdx.x;     // 0 .. N*C*C-1
    int n = idx >> 16;
    int d = (idx >> 8) & 255;
    int k = idx & 255;
    float w = W2[(n << 16) + (k << 8) + d];
    W2T[idx] = (_Float16)w;                        // W2T[n][d][k], f16 write coalesced
}

// ---------------------------------------------------------------------------
// Kernel 2: per-node fused MLP GEMM.
//   grid = (64 batch tiles of 128 rows, 64 nodes), block = 256 threads (8 waves)
//   Wave w: row slab (w>>1)*32 (two 16-row A tiles), column half (w&1)*128.
//   Each B fragment feeds TWO wmmas (row tiles share B) -> 16 B/lane/wmma LDS.
//   Accumulators: 2 x 8 tiles x v8f = 128 VGPRs (stays under 256-VGPR window).
//   W2 panels move global->LDS with double-buffered GLOBAL_LOAD_ASYNC_TO_LDS
//   in GVS mode (SGPR base + one 32-bit offset VGPR + imm offset striding both
//   LDS and global sides), fully overlapped with the previous K step's wmmas.
//   All async/LDS addresses are precomputed, asm-pinned registers -> no
//   in-loop address VALU / remat -> no WMMA->VALU WAR hazard v_nops.
//   Epilogue: relu(acc+b2) . W3, halves combined via LDS ds_add_f32, + b3.
// ---------------------------------------------------------------------------
__global__ __launch_bounds__(256)
void node_mlp_gemm_kernel(const float* __restrict__ Q,
                          const float* __restrict__ Y,
                          const float* __restrict__ W1,
                          const float* __restrict__ b1,
                          const float* __restrict__ b2,
                          const float* __restrict__ W3,
                          const float* __restrict__ b3,
                          const _Float16* __restrict__ W2T,
                          float* __restrict__ zbuf) {
    __shared__ float sW1[CDIM];
    __shared__ float sB1[CDIM];
    __shared__ float sB2[CDIM];
    __shared__ float sW3[CDIM];
    __shared__ float sZ[128];                       // per-row partial z
    __shared__ __align__(16) unsigned char sPanel[2 * PANEL_BYTES];

    const int tid     = threadIdx.x;
    const int wave    = tid >> 5;
    const int lane    = tid & 31;
    const int g       = lane >> 4;     // half-wave group (K split in frags)
    const int mlane   = lane & 15;     // A row / B col / D col within tile
    const int rowSlab = wave >> 1;     // 0..3 -> 32-row slab
    const int colHalf = wave & 1;      // 0..1 -> 128-col half
    const int n       = blockIdx.y;    // node
    const int rowBase = blockIdx.x * 128 + rowSlab * 32;
    const int row0    = rowBase + mlane;        // A-tile 0 rows
    const int row1    = row0 + 16;              // A-tile 1 rows

    // Stage per-node vectors (256 f32 each)
    sW1[tid] = W1[n * CDIM + tid];
    sB1[tid] = b1[n * CDIM + tid];
    sB2[tid] = b2[n * CDIM + tid];
    sW3[tid] = W3[n * CDIM + tid];
    if (tid < 128) sZ[tid] = 0.f;

    // Per-row scalar inputs s = q*y (both half-wave groups carry same rows)
    const float s0 = Q[row0 * NNODES + n] * Y[row0 * NNODES + n];
    const float s1 = Q[row1 * NNODES + n] * Y[row1 * NNODES + n];

    v8f acc[16];                                  // [rowTile r][colTile j]
    const v8f vzero = {0.f, 0.f, 0.f, 0.f, 0.f, 0.f, 0.f, 0.f};
#pragma unroll
    for (int t = 0; t < 16; ++t) acc[t] = vzero;

    const _Float16* w2n = W2T + (size_t)n * (CDIM * CDIM);  // uniform -> SGPRs

    // Async panel stage, GVS mode: thread tid owns panel row d=tid (64B).
    // INST_OFFSET strides BOTH the LDS dest and the global source (ISA 8 §4.4),
    // so each panel needs just one LDS-dst reg + one global-offset reg.
    unsigned ldsDst[2];
    ldsDst[0] = (unsigned)(uintptr_t)(sPanel + tid * PANEL_STRIDE);
    ldsDst[1] = ldsDst[0] + PANEL_BYTES;
    unsigned goff[8];
#pragma unroll
    for (int kb = 0; kb < 8; ++kb) {
        goff[kb] = (unsigned)((tid * CDIM + kb * 32) * 2);  // byte offset in W2T[n]
        // Pin to a VGPR (opaque to remat) so the compiler never re-derives it
        // with VALU next to a wmma (would trigger 4-NOP WAR hazard padding).
        asm volatile("" : "+v"(goff[kb]));
    }
    asm volatile("" : "+v"(ldsDst[0]), "+v"(ldsDst[1]));

    auto issue_panel = [&](int kb2, int buf) {
#pragma unroll
        for (int j = 0; j < 8; ++j) {
            asm volatile("global_load_async_to_lds_b64 %0, %1, %2 offset:%3"
                         :: "v"(ldsDst[buf]), "v"(goff[kb2]), "s"(w2n), "n"(8 * j)
                         : "memory");
        }
    };

    issue_panel(0, 0);

#pragma unroll
    for (int kb = 0; kb < 8; ++kb) {              // K steps of 32, fully unrolled
        asm volatile("s_wait_asynccnt 0x0" ::: "memory");  // panel kb landed
        __syncthreads();                          // all waves done with old buf
        if (kb < 7) issue_panel(kb + 1, (kb + 1) & 1);     // overlap with wmmas
        const unsigned char* panel = sPanel + (kb & 1) * PANEL_BYTES;

        // Two A fragments (rows share W1/b1 reads). 16-bit A 16x32 layout.
        AFrag a0, a1;
#pragma unroll
        for (int v = 0; v < 8; ++v) {
#pragma unroll
            for (int h = 0; h < 2; ++h) {
                int kl = (v < 4) ? (8 * g + 2 * v + h)
                                 : (16 + 8 * g + 2 * (v - 4) + h);
                int kk = kb * 32 + kl;
                float w1v = sW1[kk];
                float b1v = sB1[kk];
                a0.h[2 * v + h] = (_Float16)fmaxf(fmaf(s0, w1v, b1v), 0.f);
                a1.h[2 * v + h] = (_Float16)fmaxf(fmaf(s1, w1v, b1v), 0.f);
            }
        }

        // 8 column tiles in this wave's half; each B frag feeds 2 wmmas.
#pragma unroll
        for (int j = 0; j < 8; ++j) {
            BFrag bf;
            const int d = (colHalf * 8 + j) * 16 + mlane;  // output column
            const unsigned char* brow = panel + d * PANEL_STRIDE + 16 * g;
#pragma unroll
            for (int v = 0; v < 8; ++v)          // VGPR v: K = 8g + 2v (+16 for v>=4, v-4), +1
                bf.u[v] = *(const unsigned int*)(brow + 4 * v + (v >= 4 ? 16 : 0));

            acc[j] = __builtin_amdgcn_wmma_f32_16x16x32_f16(
                false, a0.v, false, bf.v, (short)0, acc[j], false, false);
            acc[8 + j] = __builtin_amdgcn_wmma_f32_16x16x32_f16(
                false, a1.v, false, bf.v, (short)0, acc[8 + j], false, false);
        }
    }

    // Epilogue: relu(acc + b2) dot W3 over this wave's 128 columns,
    // butterfly-reduce the 16-lane half-groups, combine col halves in LDS.
    // D layout: lane half holds col=mlane within tile; VGPR v -> row v + 8g.
#pragma unroll
    for (int r = 0; r < 2; ++r) {
#pragma unroll
        for (int v = 0; v < 8; ++v) {
            float p = 0.f;
#pragma unroll
            for (int j = 0; j < 8; ++j) {
                const int d = (colHalf * 8 + j) * 16 + mlane;
                float hv = acc[r * 8 + j][v] + sB2[d];
                hv = fmaxf(hv, 0.f);
                p = fmaf(hv, sW3[d], p);
            }
            p += __shfl_xor(p, 1, 32);
            p += __shfl_xor(p, 2, 32);
            p += __shfl_xor(p, 4, 32);
            p += __shfl_xor(p, 8, 32);
            if (mlane == 0) {
                const int rowLocal = rowSlab * 32 + r * 16 + v + 8 * g;
                atomicAdd(&sZ[rowLocal], p);      // ds_add_f32: combine halves
            }
        }
    }
    __syncthreads();
    if (tid < 128) {
        const float b3n = b3[n];
        zbuf[(blockIdx.x * 128 + tid) * NNODES + n] = sZ[tid] + b3n;
    }
}

// ---------------------------------------------------------------------------
// Kernel 3: z0 = bias0 - sum(s), softmax over 65 logits, relu (=identity).
// One wave per row; 8 rows per 256-thread block.
// ---------------------------------------------------------------------------
__global__ __launch_bounds__(256)
void softmax_kernel(const float* __restrict__ Q,
                    const float* __restrict__ Y,
                    const float* __restrict__ zbuf,
                    const float* __restrict__ bias0,
                    float* __restrict__ out) {
    const int row  = blockIdx.x * 8 + (threadIdx.x >> 5);
    const int lane = threadIdx.x & 31;

    // s row-sum for z0
    const float sA = Q[row * NNODES + lane]      * Y[row * NNODES + lane];
    const float sB = Q[row * NNODES + lane + 32] * Y[row * NNODES + lane + 32];
    float ssum = sA + sB;
    ssum += __shfl_xor(ssum, 1, 32);
    ssum += __shfl_xor(ssum, 2, 32);
    ssum += __shfl_xor(ssum, 4, 32);
    ssum += __shfl_xor(ssum, 8, 32);
    ssum += __shfl_xor(ssum, 16, 32);
    const float z0 = bias0[0] - ssum;

    const float z1 = zbuf[row * NNODES + lane];
    const float z2 = zbuf[row * NNODES + lane + 32];

    float mx = fmaxf(z1, z2);
    if (lane == 0) mx = fmaxf(mx, z0);
    mx = fmaxf(mx, __shfl_xor(mx, 1, 32));
    mx = fmaxf(mx, __shfl_xor(mx, 2, 32));
    mx = fmaxf(mx, __shfl_xor(mx, 4, 32));
    mx = fmaxf(mx, __shfl_xor(mx, 8, 32));
    mx = fmaxf(mx, __shfl_xor(mx, 16, 32));

    const float e1 = __expf(z1 - mx);
    const float e2 = __expf(z2 - mx);
    const float e0 = (lane == 0) ? __expf(z0 - mx) : 0.f;
    float esum = e0 + e1 + e2;
    esum += __shfl_xor(esum, 1, 32);
    esum += __shfl_xor(esum, 2, 32);
    esum += __shfl_xor(esum, 4, 32);
    esum += __shfl_xor(esum, 8, 32);
    esum += __shfl_xor(esum, 16, 32);
    const float inv = 1.f / esum;

    float* orow = out + (size_t)row * (NNODES + 1);
    if (lane == 0) orow[0] = e0 * inv;
    orow[1 + lane]  = e1 * inv;       // softmax outputs >= 0; relu = identity
    orow[33 + lane] = e2 * inv;
}

// ---------------------------------------------------------------------------
extern "C" void kernel_launch(void* const* d_in, const int* in_sizes, int n_in,
                              void* d_out, int out_size, void* d_ws, size_t ws_size,
                              hipStream_t stream) {
    const float* Q     = (const float*)d_in[0];
    const float* Y     = (const float*)d_in[1];
    const float* W1    = (const float*)d_in[2];
    const float* b1    = (const float*)d_in[3];
    const float* W2    = (const float*)d_in[4];
    const float* b2    = (const float*)d_in[5];
    const float* W3    = (const float*)d_in[6];
    const float* b3    = (const float*)d_in[7];
    const float* bias0 = (const float*)d_in[8];
    float* out = (float*)d_out;

    // Workspace layout: W2T f16 [N*C*C] (8 MB), zbuf f32 [B*N] (2 MB)
    _Float16* W2T = (_Float16*)d_ws;
    float* zbuf = (float*)((char*)d_ws + (size_t)NNODES * CDIM * CDIM * sizeof(_Float16));

    // 1) W2 -> f16, transposed per node to [n][d][k]
    prep_w2_kernel<<<(NNODES * CDIM * CDIM) / 256, 256, 0, stream>>>(W2, W2T);

    // 2) per-node WMMA GEMM: grid (batch tiles, nodes)
    dim3 grid(BDIM / 128, NNODES);
    node_mlp_gemm_kernel<<<grid, 256, 0, stream>>>(Q, Y, W1, b1, b2, W3, b3, W2T, zbuf);

    // 3) z0 + softmax + relu
    softmax_kernel<<<BDIM / 8, 256, 0, stream>>>(Q, Y, zbuf, bias0, out);
}
